// MultiDomainSparseAttention_16045997817876
// MI455X (gfx1250) — matrix operand order repaired and checked by reading, not compiled, not run on hardware
//
#include <hip/hip_runtime.h>
#include <hip/hip_bf16.h>

// MultiDomainSparseAttention for MI455X (gfx1250, wave32, WMMA + async-to-LDS).
// N=4, C=256, L=4096, Cr=64, NH=4, CHUNK=64, HB=64.

#define N_      4
#define C_      256
#define L_      4096
#define LP_     4112            // 8 + L + 8 padded f16 row (8224 B, 16B aligned)
#define CR_     64
#define NH_     4
#define NHL_    (NH_ * L_)      // 16384
#define CHUNK_  64
#define NCHUNK_ (L_ / CHUNK_)   // 64
#define HBH_    32              // HB/2
#define EPS_BN   1e-5f
#define EPS_NORM 5e-5f

typedef __attribute__((ext_vector_type(16))) _Float16 v16h;
typedef __attribute__((ext_vector_type(8)))  _Float16 h8;
typedef __attribute__((ext_vector_type(8)))  float    v8f;

__device__ __forceinline__ v8f wmma_f16f32(v16h a, v16h b, v8f c) {
  // D = A(16x32 f16) * B(32x16 f16) + C(16x16 f32)
  return __builtin_amdgcn_wmma_f32_16x16x32_f16(false, a, false, b, (short)0, c,
                                                false, false);
}

// CDNA5 async global->LDS copy (ASYNCcnt tracked), 16B per lane.
// VDST = VGPR holding LDS byte address (low 32 bits of generic shared ptr),
// VADDR = 64-bit global address. GV mode (saddr = off).
__device__ __forceinline__ void async_b128(unsigned lds_addr, const void* gsrc) {
  asm volatile("global_load_async_to_lds_b128 %0, %1, off"
               :: "v"(lds_addr), "v"(gsrc)
               : "memory");
}
__device__ __forceinline__ void wait_async0() {
  asm volatile("s_wait_asynccnt 0" ::: "memory");
}
__device__ __forceinline__ unsigned lds_off(const void* p) {
  return (unsigned)(uintptr_t)p;  // LDS aperture: addr[31:0] is the LDS address
}

// A fragment: 16x32 f16, row-major source with leading dim lda.
// ISA 7.12.2: lane m = lane%16, half = lane/16;
// VGPR j holds K = (j%4)*2 + (j/4)*16 + half*8 (+1).
__device__ __forceinline__ v16h frag_a_f16(const _Float16* A, int lda, int lane) {
  int m = lane & 15, half = lane >> 4;
  v16h r;
#pragma unroll
  for (int j = 0; j < 8; ++j) {
    int k0 = ((j & 3) << 1) + ((j >> 2) << 4) + (half << 3);
    r[2 * j]     = A[m * lda + k0];
    r[2 * j + 1] = A[m * lda + k0 + 1];
  }
  return r;
}

__device__ __forceinline__ v16h frag_a_f32(const float* A, int lda, int lane) {
  int m = lane & 15, half = lane >> 4;
  v16h r;
#pragma unroll
  for (int j = 0; j < 8; ++j) {
    int k0 = ((j & 3) << 1) + ((j >> 2) << 4) + (half << 3);
    r[2 * j]     = (_Float16)A[m * lda + k0];
    r[2 * j + 1] = (_Float16)A[m * lda + k0 + 1];
  }
  return r;
}

// B fragment: 32x16 f16 (KxN), row-major source (ldb = N stride).
// Lane supplies N=lane%16; VGPR j holds K = (j%4)*2 + (j/4)*16 + half*8 (+1).
__device__ __forceinline__ v16h frag_b_f16(const _Float16* B, int ldb, int lane) {
  int nn = lane & 15, half = lane >> 4;
  v16h r;
#pragma unroll
  for (int j = 0; j < 8; ++j) {
    int k0 = ((j & 3) << 1) + ((j >> 2) << 4) + (half << 3);
    r[2 * j]     = B[k0 * ldb + nn];
    r[2 * j + 1] = B[(k0 + 1) * ldb + nn];
  }
  return r;
}

// ---------------------------------------------------------------------------
// Prep 0a: x [N,C,L] f32 -> padded f16 [N,C,LP_] (zeros in pad) for async GEMMs
// ---------------------------------------------------------------------------
__global__ __launch_bounds__(256) void k_prep_x(const float* __restrict__ x,
                                                _Float16* __restrict__ xp) {
  int i = blockIdx.x * 256 + threadIdx.x;
  if (i >= N_ * C_ * LP_) return;
  int lp = i % LP_;
  int nc = i / LP_;
  float v = (lp >= 8 && lp < 8 + L_) ? x[(size_t)nc * L_ + (lp - 8)] : 0.f;
  xp[i] = (_Float16)v;
}

// Prep 0b: weights -> f16. w_asm [C,C] kept; w_match -> per-tap [3][Cr][C].
__global__ __launch_bounds__(256) void k_prep_w(const float* __restrict__ wasm,
                                                const float* __restrict__ wm,
                                                _Float16* __restrict__ wa16,
                                                _Float16* __restrict__ wm16) {
  int i = blockIdx.x * 256 + threadIdx.x;
  if (i < C_ * C_) wa16[i] = (_Float16)wasm[i];
  if (i < 3 * CR_ * C_) {
    int tap = i / (CR_ * C_);
    int r = i % (CR_ * C_);
    int cr = r / C_, c = r % C_;
    wm16[i] = (_Float16)wm[((size_t)cr * C_ + c) * 3 + tap];
  }
}

// ---------------------------------------------------------------------------
// Kernel 1: conv_assembly (1x1) + BN2 + ReLU -> y_embed f16 [N,L,C]
// GEMM (flipped): D[co, l] = sum_c W[co,c] * x[n,c,l]
// A = W tile [64 co x 32 c] (rows contiguous), B = x tile [32 c x 64 l]
// (rows contiguous). Both staged by global_load_async_to_lds_b128.
// ---------------------------------------------------------------------------
__global__ __launch_bounds__(512) void k_conv_asm(
    const _Float16* __restrict__ xp, const _Float16* __restrict__ w16,
    const float* __restrict__ g, const float* __restrict__ bb,
    const float* __restrict__ mm, const float* __restrict__ vv,
    _Float16* __restrict__ ye) {
  __shared__ _Float16 Aw[64 * 32];  // [co][c]
  __shared__ _Float16 Bx[32 * 64];  // [c][l]
  int n = blockIdx.z, l0 = blockIdx.x * 64, co0 = blockIdx.y * 64;
  int tid = threadIdx.x, lane = tid & 31, wave = tid >> 5;
  int mt = wave >> 2, nt = wave & 3;  // mt: co tile, nt: l tile
  v8f acc = {0.f, 0.f, 0.f, 0.f, 0.f, 0.f, 0.f, 0.f};
  const _Float16* xbase = xp + (size_t)n * C_ * LP_ + 8 + l0;
  for (int kk = 0; kk < C_; kk += 32) {
    if (tid < 256) {  // A: 64 rows x 4 segs of 8 f16 (16B)
      int row = tid >> 2, seg = tid & 3;
      async_b128(lds_off(&Aw[row * 32 + seg * 8]),
                 w16 + (size_t)(co0 + row) * C_ + kk + seg * 8);
    } else {          // B: 32 rows x 8 segs of 8 f16
      int j = tid - 256;
      int row = j >> 3, seg = j & 7;
      async_b128(lds_off(&Bx[row * 64 + seg * 8]),
                 xbase + (size_t)(kk + row) * LP_ + seg * 8);
    }
    wait_async0();
    __syncthreads();
    v16h a = frag_a_f16(Aw + mt * 16 * 32, 32, lane);
    v16h b = frag_b_f16(Bx + nt * 16, 64, lane);
    acc = wmma_f16f32(a, b, acc);
    __syncthreads();
  }
  // D: rows = co, cols = l. Per lane: 8 consecutive co -> one 16B store.
  int l = l0 + nt * 16 + (lane & 15);
  int cb = co0 + mt * 16 + 8 * (lane >> 4);
  h8 o;
#pragma unroll
  for (int vr = 0; vr < 8; ++vr) {
    int co = cb + vr;
    float sc = g[co] * rsqrtf(vv[co] + EPS_BN);
    float sh = bb[co] - mm[co] * sc;
    o[vr] = (_Float16)fmaxf(acc[vr] * sc + sh, 0.f);
  }
  *(h8*)(ye + (size_t)(n * L_ + l) * C_ + cb) = o;
}

// ---------------------------------------------------------------------------
// Kernel 2: conv_match (3-tap, pad 1) + BN1 + ReLU -> x_embed f32 [N,L,Cr]
// D[cr, l] += sum_c Wtap[cr,c] * x[n,c,l+tap-1]; taps share one staged
// 80-wide x window (shift = LDS base offset; zero-pad lives in xp).
// ---------------------------------------------------------------------------
__global__ __launch_bounds__(512) void k_conv_match(
    const _Float16* __restrict__ xp, const _Float16* __restrict__ wm16,
    const float* __restrict__ g, const float* __restrict__ bb,
    const float* __restrict__ mm, const float* __restrict__ vv,
    float* __restrict__ xe) {
  __shared__ _Float16 Aw[3 * 64 * 32];  // [tap][cr][c]  12 KB
  __shared__ _Float16 Bx[32 * 80];      // [c][l0-8 .. l0+71]  5 KB
  int n = blockIdx.z, l0 = blockIdx.x * 64;
  int tid = threadIdx.x, lane = tid & 31, wave = tid >> 5;
  int mt = wave >> 2, nt = wave & 3;  // mt: cr tile, nt: l tile
  v8f acc = {0.f, 0.f, 0.f, 0.f, 0.f, 0.f, 0.f, 0.f};
  const _Float16* xbase = xp + (size_t)n * C_ * LP_ + l0;  // element l0-8
  for (int kk = 0; kk < C_; kk += 32) {
    for (int i = tid; i < 1088; i += 512) {
      if (i < 768) {  // A: 3 taps x 64 rows x 4 segs
        int tap = i >> 8, j = i & 255;
        int row = j >> 2, seg = j & 3;
        async_b128(lds_off(&Aw[tap * 2048 + row * 32 + seg * 8]),
                   wm16 + (size_t)tap * CR_ * C_ + (size_t)row * C_ + kk +
                       seg * 8);
      } else {        // B: 32 rows x 10 segs of 8 f16 (80-wide window)
        int j = i - 768;
        int row = j / 10, seg = j % 10;
        async_b128(lds_off(&Bx[row * 80 + seg * 8]),
                   xbase + (size_t)(kk + row) * LP_ + seg * 8);
      }
    }
    wait_async0();
    __syncthreads();
#pragma unroll
    for (int tap = 0; tap < 3; ++tap) {
      v16h a = frag_a_f16(Aw + tap * 2048 + mt * 16 * 32, 32, lane);
      // col for output l index q is q + tap - 1 relative to l0 => Bx col q+tap+7
      v16h b = frag_b_f16(Bx + nt * 16 + tap + 7, 80, lane);
      acc = wmma_f16f32(a, b, acc);
    }
    __syncthreads();
  }
  int l = l0 + nt * 16 + (lane & 15);
  int crb = mt * 16 + 8 * (lane >> 4);
  float* xrow = xe + (size_t)(n * L_ + l) * CR_ + crb;
#pragma unroll
  for (int vr = 0; vr < 8; ++vr) {
    int cr = crb + vr;
    float sc = g[cr] * rsqrtf(vv[cr] + EPS_BN);
    float sh = bb[cr] - mm[cr] * sc;
    xrow[vr] = fmaxf(acc[vr] * sc + sh, 0.f);
  }
}

// ---------------------------------------------------------------------------
// Kernel 3: LSH hash codes. rv = x_embed @ rot ; code = argmax([rv, -rv]) + h
// ---------------------------------------------------------------------------
__global__ __launch_bounds__(256) void k_hash(
    const float* __restrict__ xe, const float* __restrict__ rot,
    int* __restrict__ codes) {
  __shared__ float rotS[CR_ * NH_ * HBH_];  // 32 KB
  int tid = threadIdx.x;
  for (int i = tid; i < CR_ * NH_ * HBH_; i += 256) rotS[i] = rot[i];
  __syncthreads();
  int gid = blockIdx.x * 256 + tid;
  int n = gid / L_, t = gid % L_;
  float xr[CR_];
#pragma unroll 8
  for (int f = 0; f < CR_; ++f) xr[f] = xe[(size_t)(n * L_ + t) * CR_ + f];
#pragma unroll
  for (int h = 0; h < NH_; ++h) {
    float acc[HBH_];
#pragma unroll
    for (int i = 0; i < HBH_; ++i) acc[i] = 0.f;
#pragma unroll 4
    for (int f = 0; f < CR_; ++f) {
      float xv = xr[f];
      const float* rp = rotS + (f * NH_ + h) * HBH_;
#pragma unroll
      for (int i = 0; i < HBH_; ++i) acc[i] += xv * rp[i];
    }
    float bp = -1e30f, bn = -1e30f;
    int ip = 0, in2 = 0;
#pragma unroll
    for (int i = 0; i < HBH_; ++i) {
      if (acc[i] > bp) { bp = acc[i]; ip = i; }
      if (-acc[i] > bn) { bn = -acc[i]; in2 = i; }
    }
    int code = (bp >= bn) ? ip : (HBH_ + in2);
    codes[n * NHL_ + h * L_ + t] = code + h;
  }
}

// ---------------------------------------------------------------------------
// Kernels 4-7: stable counting sort (keys 0..66 < 128) == stable argsort
// ---------------------------------------------------------------------------
__global__ void k_zero_hist(int* __restrict__ hist) {
  int i = blockIdx.x * blockDim.x + threadIdx.x;
  if (i < N_ * 128) hist[i] = 0;
}

__global__ __launch_bounds__(256) void k_hist(const int* __restrict__ codes,
                                              int* __restrict__ hist) {
  int gid = blockIdx.x * 256 + threadIdx.x;
  int n = gid / NHL_;
  atomicAdd(&hist[n * 128 + codes[gid]], 1);
}

__global__ void k_scan(const int* __restrict__ hist, int* __restrict__ offs) {
  if (threadIdx.x == 0) {
    int n = blockIdx.x, s = 0;
    for (int kkey = 0; kkey < 128; ++kkey) {
      offs[n * 128 + kkey] = s;
      s += hist[n * 128 + kkey];
    }
  }
}

__global__ __launch_bounds__(128) void k_scatter(
    const int* __restrict__ codes, const int* __restrict__ offs,
    int* __restrict__ indices, int* __restrict__ undo) {
  int n = blockIdx.x, key = threadIdx.x;
  int off = offs[n * 128 + key];
  const int* cp = codes + (size_t)n * NHL_;
  for (int s = 0; s < NHL_; ++s) {
    if (cp[s] == key) {
      indices[(size_t)n * NHL_ + off] = s;
      undo[(size_t)n * NHL_ + s] = off;
      ++off;
    }
  }
}

// ---------------------------------------------------------------------------
// Kernel 8: gather sorted rows + L2-normalize (wave32 shfl reduce) -> f16
// ---------------------------------------------------------------------------
__global__ __launch_bounds__(256) void k_gather(
    const float* __restrict__ xe, const _Float16* __restrict__ ye,
    const int* __restrict__ indices, _Float16* __restrict__ xb,
    _Float16* __restrict__ xm, _Float16* __restrict__ ys) {
  int tid = threadIdx.x, lane = tid & 31, wv = tid >> 5;
  int row = blockIdx.x * 8 + wv;  // 0 .. N*NHL-1
  int n = row >> 14;              // NHL == 16384
  int idx = indices[row];
  int t = idx & (L_ - 1);         // indices % L
  const float* xp = xe + (size_t)(n * L_ + t) * CR_;
  float x0 = xp[lane], x1 = xp[lane + 32];
  float ss = x0 * x0 + x1 * x1;
#pragma unroll
  for (int d = 16; d >= 1; d >>= 1) ss += __shfl_xor(ss, d, 32);
  float inv = 1.f / fmaxf(sqrtf(ss), EPS_NORM);
  xb[(size_t)row * CR_ + lane]      = (_Float16)x0;
  xb[(size_t)row * CR_ + lane + 32] = (_Float16)x1;
  xm[(size_t)row * CR_ + lane]      = (_Float16)(x0 * inv);
  xm[(size_t)row * CR_ + lane + 32] = (_Float16)(x1 * inv);
  const _Float16* yp = ye + (size_t)(n * L_ + t) * C_;
  _Float16* yd = ys + (size_t)row * C_;
#pragma unroll
  for (int j = 0; j < 8; ++j) yd[lane + 32 * j] = yp[lane + 32 * j];
}

// ---------------------------------------------------------------------------
// Kernel 9: chunked attention per (n,h,chunk): S = Q*K^T (64x192), exact
// logsumexp softmax (lse stored in SORTED order, faithful to reference),
// O = P*V (64x256). Q/K staged via async-to-LDS; V streamed from L2
// (entire ~130MB working set fits MI455X's 192MB L2).
// ---------------------------------------------------------------------------
__device__ __forceinline__ int chunk3(int third, int k) {
  return (third == 0) ? k
         : (third == 1) ? ((k + NCHUNK_ - 1) & (NCHUNK_ - 1))
                        : ((k + 1) & (NCHUNK_ - 1));
}

__global__ __launch_bounds__(256) void k_attn(
    const _Float16* __restrict__ xb, const _Float16* __restrict__ xm,
    const _Float16* __restrict__ ys, float* __restrict__ att,
    float* __restrict__ bscore) {
  __shared__ _Float16 sK[192 * 64];  // 24 KB
  __shared__ _Float16 sQ[32 * 64];   //  4 KB
  __shared__ float    sS[32 * 192];  // 24 KB
  int blk = blockIdx.x;  // n*NH*64 + h*64 + k
  int k = blk & 63;
  int h = (blk >> 6) & 3;
  int n = blk >> 8;
  int tid = threadIdx.x, lane = tid & 31, wave = tid >> 5;
  size_t headBase = (size_t)n * NHL_ + (size_t)h * L_;

  // Async-stage K window: chunks [k, k-1, k+1] of normalized rows.
  for (int i = tid; i < 1536; i += 256) {  // 192 rows x 8 segs of 16B
    int row = i >> 3, seg = i & 7;
    int ch = chunk3(row >> 6, k);
    async_b128(lds_off(&sK[row * 64 + seg * 8]),
               xm + (headBase + ch * CHUNK_ + (row & 63)) * CR_ + seg * 8);
  }
  wait_async0();
  __syncthreads();

  for (int half = 0; half < 2; ++half) {
    int row0 = k * CHUNK_ + half * 32;  // sorted row within head
    {  // async-stage half-chunk of Q: 32 rows x 8 segs = 256 ops
      int row = tid >> 3, seg = tid & 7;
      async_b128(lds_off(&sQ[row * 64 + seg * 8]),
                 xb + (headBase + row0 + row) * CR_ + seg * 8);
    }
    wait_async0();
    __syncthreads();

    // Scores: 2x12 tiles of 16x16 (K-dim 64), 3 per wave (8 waves).
#pragma unroll
    for (int q = 0; q < 3; ++q) {
      int t = wave * 3 + q;
      int mt = t / 12, ct = t % 12;
      v8f acc = {0.f, 0.f, 0.f, 0.f, 0.f, 0.f, 0.f, 0.f};
#pragma unroll
      for (int kk = 0; kk < 64; kk += 32) {
        v16h a = frag_a_f16(sQ + mt * 16 * 64 + kk, 64, lane);
        v16h b;  // B[kq][nj] = sK[(ct*16+nj)][kk+kq]
#pragma unroll
        for (int j = 0; j < 8; ++j) {
          int k0 = ((j & 3) << 1) + ((j >> 2) << 4) + ((lane >> 4) << 3);
          b[2 * j]     = sK[(ct * 16 + (lane & 15)) * 64 + kk + k0];
          b[2 * j + 1] = sK[(ct * 16 + (lane & 15)) * 64 + kk + k0 + 1];
        }
        acc = wmma_f16f32(a, b, acc);
      }
      int col = ct * 16 + (lane & 15);
#pragma unroll
      for (int vr = 0; vr < 8; ++vr)
        sS[(mt * 16 + vr + 8 * (lane >> 4)) * 192 + col] = acc[vr];
    }
    __syncthreads();

    // Exact row softmax with logsumexp (32 rows, one lane each).
    if (tid < 32) {
      float mx = -1e30f;
      for (int j = 0; j < 192; ++j) mx = fmaxf(mx, sS[tid * 192 + j]);
      float sum = 0.f;
      for (int j = 0; j < 192; ++j) sum += __expf(sS[tid * 192 + j] - mx);
      float lse = mx + __logf(sum);
      bscore[headBase + row0 + tid] = lse;  // NOT undo-sorted (faithful)
      for (int j = 0; j < 192; ++j)
        sS[tid * 192 + j] = __expf(sS[tid * 192 + j] - lse);
    }
    __syncthreads();

    // Output: 2x16 tiles of 16x16 (K-dim 192), 4 per wave.
#pragma unroll
    for (int q = 0; q < 4; ++q) {
      int t = wave * 4 + q;
      int mt = t >> 4, ct = t & 15;
      v8f acc = {0.f, 0.f, 0.f, 0.f, 0.f, 0.f, 0.f, 0.f};
      for (int kk = 0; kk < 192; kk += 32) {
        v16h a = frag_a_f32(sS + (mt * 16) * 192 + kk, 192, lane);
        v16h b;
#pragma unroll
        for (int jj = 0; jj < 8; ++jj) {
          int j = kk + ((jj & 3) << 1) + ((jj >> 2) << 4) + ((lane >> 4) << 3);
          const _Float16* vp =
              ys + (headBase + chunk3(j >> 6, k) * CHUNK_ + (j & 63)) * C_ +
              ct * 16 + (lane & 15);
          b[2 * jj]     = vp[0];
          b[2 * jj + 1] = vp[C_];
        }
        acc = wmma_f16f32(a, b, acc);
      }
      int col = ct * 16 + (lane & 15);
#pragma unroll
      for (int vr = 0; vr < 8; ++vr) {
        int mrow = mt * 16 + vr + 8 * (lane >> 4);
        att[(headBase + row0 + mrow) * C_ + col] = acc[vr];
      }
    }
    __syncthreads();
  }
}

// ---------------------------------------------------------------------------
// Kernel 10: undo-sort gather + head softmax + BN3 + residual -> out [N,C,L]
// ---------------------------------------------------------------------------
__global__ __launch_bounds__(256) void k_combine(
    const float* __restrict__ att, const float* __restrict__ bscore,
    const int* __restrict__ undo, const float* __restrict__ x,
    const float* __restrict__ g, const float* __restrict__ bb,
    const float* __restrict__ mm, const float* __restrict__ vv,
    float* __restrict__ out) {
  __shared__ float sb[NH_];
  __shared__ int su[NH_];
  int blk = blockIdx.x;
  int n = blk >> 12, l = blk & (L_ - 1);
  int tid = threadIdx.x;
  if (tid < NH_) {
    sb[tid] = bscore[(size_t)n * NHL_ + tid * L_ + l];
    su[tid] = undo[(size_t)n * NHL_ + tid * L_ + l];
  }
  __syncthreads();
  float mx = fmaxf(fmaxf(sb[0], sb[1]), fmaxf(sb[2], sb[3]));
  float e0 = __expf(sb[0] - mx), e1 = __expf(sb[1] - mx);
  float e2 = __expf(sb[2] - mx), e3 = __expf(sb[3] - mx);
  float es = e0 + e1 + e2 + e3;
  float p[NH_] = {e0 / es, e1 / es, e2 / es, e3 / es};
  int c = tid;
  float acc = 0.f;
#pragma unroll
  for (int h = 0; h < NH_; ++h)
    acc += att[((size_t)n * NHL_ + su[h]) * C_ + c] * p[h];
  float sc = g[c] * rsqrtf(vv[c] + EPS_BN);
  float sh = bb[c] - mm[c] * sc;
  float y = acc * sc + sh + x[(size_t)n * C_ * L_ + (size_t)c * L_ + l];
  out[(size_t)n * C_ * L_ + (size_t)c * L_ + l] = y;
}

// ---------------------------------------------------------------------------
extern "C" void kernel_launch(void* const* d_in, const int* in_sizes, int n_in,
                              void* d_out, int out_size, void* d_ws,
                              size_t ws_size, hipStream_t stream) {
  const float* x    = (const float*)d_in[0];
  const float* rot  = (const float*)d_in[1];
  const float* wm   = (const float*)d_in[2];
  const float* bn1g = (const float*)d_in[3];
  const float* bn1b = (const float*)d_in[4];
  const float* bn1m = (const float*)d_in[5];
  const float* bn1v = (const float*)d_in[6];
  const float* wasm = (const float*)d_in[7];
  const float* bn2g = (const float*)d_in[8];
  const float* bn2b = (const float*)d_in[9];
  const float* bn2m = (const float*)d_in[10];
  const float* bn2v = (const float*)d_in[11];
  const float* bn3g = (const float*)d_in[12];
  const float* bn3b = (const float*)d_in[13];
  const float* bn3m = (const float*)d_in[14];
  const float* bn3v = (const float*)d_in[15];

  // Workspace carve (256B aligned). Total ~136 MB.
  char* ws = (char*)d_ws;
  auto carve = [&](size_t bytes) -> char* {
    char* p = ws;
    ws += (bytes + 255) & ~(size_t)255;
    return p;
  };
  _Float16* xp      = (_Float16*)carve((size_t)N_ * C_ * LP_ * 2);   // 8.4 MB
  _Float16* wa16    = (_Float16*)carve((size_t)C_ * C_ * 2);
  _Float16* wm16    = (_Float16*)carve((size_t)3 * CR_ * C_ * 2);
  float*    xe      = (float*)carve((size_t)N_ * L_ * CR_ * 4);      //  4 MB
  _Float16* ye      = (_Float16*)carve((size_t)N_ * L_ * C_ * 2);    //  8 MB
  int*      codes   = (int*)carve((size_t)N_ * NHL_ * 4);            // 256 KB
  int*      hist    = (int*)carve((size_t)N_ * 128 * 4);
  int*      offs    = (int*)carve((size_t)N_ * 128 * 4);
  int*      indices = (int*)carve((size_t)N_ * NHL_ * 4);
  int*      undo    = (int*)carve((size_t)N_ * NHL_ * 4);
  _Float16* xb      = (_Float16*)carve((size_t)N_ * NHL_ * CR_ * 2); //  8 MB
  _Float16* xm      = (_Float16*)carve((size_t)N_ * NHL_ * CR_ * 2); //  8 MB
  _Float16* ys      = (_Float16*)carve((size_t)N_ * NHL_ * C_ * 2);  // 32 MB
  float*    bscore  = (float*)carve((size_t)N_ * NHL_ * 4);          // 256 KB
  float*    att     = (float*)carve((size_t)N_ * NHL_ * C_ * 4);     // 64 MB

  k_prep_x<<<dim3((N_ * C_ * LP_ + 255) / 256), 256, 0, stream>>>(x, xp);
  k_prep_w<<<dim3((C_ * C_ + 255) / 256), 256, 0, stream>>>(wasm, wm, wa16,
                                                            wm16);
  k_conv_asm<<<dim3(L_ / 64, C_ / 64, N_), 512, 0, stream>>>(
      xp, wa16, bn2g, bn2b, bn2m, bn2v, ye);
  k_conv_match<<<dim3(L_ / 64, 1, N_), 512, 0, stream>>>(
      xp, wm16, bn1g, bn1b, bn1m, bn1v, xe);
  k_hash<<<dim3((N_ * L_) / 256), 256, 0, stream>>>(xe, rot, codes);
  k_zero_hist<<<1, 512, 0, stream>>>(hist);
  k_hist<<<dim3((N_ * NHL_) / 256), 256, 0, stream>>>(codes, hist);
  k_scan<<<N_, 32, 0, stream>>>(hist, offs);
  k_scatter<<<N_, 128, 0, stream>>>(codes, offs, indices, undo);
  k_gather<<<dim3((N_ * NHL_) / 8), 256, 0, stream>>>(xe, ye, indices, xb, xm,
                                                      ys);
  k_attn<<<dim3(N_ * NH_ * NCHUNK_), 256, 0, stream>>>(xb, xm, ys, att,
                                                       bscore);
  k_combine<<<dim3(N_ * L_), 256, 0, stream>>>(att, bscore, undo, x, bn3g,
                                               bn3b, bn3m, bn3v,
                                               (float*)d_out);
  (void)in_sizes; (void)n_in; (void)out_size; (void)ws_size;
}
